// PointCloudTransformerEncoder_71038759075841
// MI455X (gfx1250) — matrix operand order repaired and checked by reading, not compiled, not run on hardware
//
#include <hip/hip_runtime.h>
#include <stdint.h>

// ---------------------------------------------------------------------------
// CDNA5 (gfx1250) vector types for WMMA
// ---------------------------------------------------------------------------
typedef __attribute__((ext_vector_type(16))) __bf16 v16bf;
typedef __attribute__((ext_vector_type(8)))  __bf16 v8bf;
typedef __attribute__((ext_vector_type(8)))  float  v8f;

// ---------------------------------------------------------------------------
// Device helpers
// ---------------------------------------------------------------------------
__device__ __forceinline__ v8f wmma_bf16(v16bf a, v16bf b, v8f c) {
  // D = A(16x32) * B(32x16) + C, fp32 accumulate
  return __builtin_amdgcn_wmma_f32_16x16x32_bf16(
      /*neg_a=*/false, a, /*neg_b=*/false, b,
      /*c_mod=*/(short)0, c, /*reuse_a=*/false, /*reuse_b=*/false);
}

// Build a 16-half fragment from two contiguous 8-half (16B) chunks.
__device__ __forceinline__ v16bf frag_from2(const __bf16* p0, const __bf16* p1) {
  v8bf lo = *(const v8bf*)p0;
  v8bf hi = *(const v8bf*)p1;
  v16bf r;
#pragma unroll
  for (int i = 0; i < 8; ++i) { r[i] = lo[i]; r[i + 8] = hi[i]; }
  return r;
}

// gfx1250 async global->LDS copy (16B per lane), tracked by ASYNCcnt.
__device__ __forceinline__ void async_copy16(void* lds, const void* g) {
  unsigned l = (unsigned)(size_t)lds;          // generic->LDS byte offset (low 32b)
  unsigned long long a = (unsigned long long)(size_t)g;
  asm volatile("global_load_async_to_lds_b128 %0, %1, off"
               :: "v"(l), "v"(a) : "memory");
}
__device__ __forceinline__ void async_wait0() {
  asm volatile("s_wait_asynccnt 0" ::: "memory");
}

// ---------------------------------------------------------------------------
// Weight convert + transpose: dst[n*K + k] = (bf16) src[k*N + n]
// ---------------------------------------------------------------------------
__global__ void k_convT(const float* __restrict__ src, __bf16* __restrict__ dst,
                        int K, int N) {
  int idx = blockIdx.x * blockDim.x + threadIdx.x;
  if (idx >= K * N) return;
  int k = idx / N, n = idx % N;
  dst[(size_t)n * K + k] = (__bf16)src[idx];
}

// ---------------------------------------------------------------------------
// Embed: out[row, c] = b0[c] + sum_i in[row, i] * w0[i, c]   (IN = 3)
// ---------------------------------------------------------------------------
__global__ void k_embed(const float* __restrict__ in, const float* __restrict__ w0,
                        const float* __restrict__ b0, float* __restrict__ out,
                        int total, int C) {
  int idx = blockIdx.x * blockDim.x + threadIdx.x;
  if (idx >= total) return;
  int row = idx / C, c = idx % C;
  const float* xr = in + (size_t)row * 3;
  out[idx] = b0[c] + xr[0] * w0[c] + xr[1] * w0[C + c] + xr[2] * w0[2 * C + c];
}

// ---------------------------------------------------------------------------
// Point-axis LayerNorm stats: per (b,c) over N points. blocks = B*C.
// ---------------------------------------------------------------------------
__global__ __launch_bounds__(256)
void k_colstats(const float* __restrict__ x, float* __restrict__ mean,
                float* __restrict__ rstd, int Npts, int C) {
  __shared__ float s1[256], s2[256];
  int bc = blockIdx.x;
  int b = bc / C, c = bc % C;
  const float* base = x + (size_t)b * Npts * C + c;
  float a = 0.f, q = 0.f;
  for (int n = threadIdx.x; n < Npts; n += 256) {
    float v = base[(size_t)n * C];
    a += v; q += v * v;
  }
  s1[threadIdx.x] = a; s2[threadIdx.x] = q;
  __syncthreads();
  for (int st = 128; st > 0; st >>= 1) {
    if ((int)threadIdx.x < st) {
      s1[threadIdx.x] += s1[threadIdx.x + st];
      s2[threadIdx.x] += s2[threadIdx.x + st];
    }
    __syncthreads();
  }
  if (threadIdx.x == 0) {
    float mu = s1[0] / (float)Npts;
    float var = s2[0] / (float)Npts - mu * mu;
    mean[bc] = mu;
    rstd[bc] = rsqrtf(var + 1e-6f);
  }
}

// ---------------------------------------------------------------------------
// LN apply (+optional relu, +optional residual), write f32 / bf16 / concat slice
// ---------------------------------------------------------------------------
__global__ void k_ln_apply(const float* __restrict__ x, const float* __restrict__ res,
                           const float* __restrict__ mean, const float* __restrict__ rstd,
                           const float* __restrict__ sc, const float* __restrict__ bs,
                           float* __restrict__ outF, __bf16* __restrict__ outB,
                           __bf16* __restrict__ outB2, int ostride,
                           int total, int NC, int C, int relu) {
  int idx = blockIdx.x * blockDim.x + threadIdx.x;
  if (idx >= total) return;
  int c = idx % C;
  int bc = (idx / NC) * C + c;
  float v = (x[idx] - mean[bc]) * rstd[bc] * sc[c] + bs[c];
  if (relu) v = fmaxf(v, 0.f);
  if (res) v += res[idx];
  if (outF) outF[idx] = v;
  if (outB) outB[idx] = (__bf16)v;
  if (outB2) { int row = idx / C; outB2[(size_t)row * ostride + c] = (__bf16)v; }
}

__global__ void k_zero(float* __restrict__ p, int n) {
  int idx = blockIdx.x * blockDim.x + threadIdx.x;
  if (idx < n) p[idx] = 0.f;
}

// ---------------------------------------------------------------------------
// Generic bf16 WMMA GEMM: Y[M,Nout] = X[M,K] @ Wt[Nout,K]^T + bias
// Block tile 128x64, 8 waves (each wave: 32x32 = 2x2 WMMA tiles), K step 32.
// A/B tiles staged in LDS via async global->LDS copies.
// ---------------------------------------------------------------------------
__global__ __launch_bounds__(256)
void k_gemm_bf16(const __bf16* __restrict__ X, const __bf16* __restrict__ Wt,
                 const float* __restrict__ bias,
                 float* __restrict__ outF, __bf16* __restrict__ outB,
                 int M, int Kd, int Nout) {
  __shared__ __bf16 Al[128 * 32];
  __shared__ __bf16 Bl[64 * 32];
  const int m0 = blockIdx.x * 128, n0 = blockIdx.y * 64;
  const int tid = threadIdx.x, lane = tid & 31, wid = tid >> 5;
  const int h = lane >> 4;                 // lane half selects K sub-run
  const int wm = wid & 3, wn = wid >> 2;   // 4x2 wave grid over 128x64 tile
  v8f acc[2][2] = {};

  for (int k0 = 0; k0 < Kd; k0 += 32) {
    __syncthreads();  // protect LDS reads of previous iteration
    // stage A: 128 rows x 32 cols bf16 (8KB) -> 512 x 16B chunks
    for (int c = tid; c < 512; c += 256) {
      int r = c >> 2, p = c & 3;
      async_copy16(&Al[r * 32 + p * 8], X + (size_t)(m0 + r) * Kd + k0 + p * 8);
    }
    // stage B (pre-transposed weights): 64 rows x 32 cols (4KB) -> 256 chunks
    {
      int r = tid >> 2, p = tid & 3;
      async_copy16(&Bl[r * 32 + p * 8], Wt + (size_t)(n0 + r) * Kd + k0 + p * 8);
    }
    async_wait0();
    __syncthreads();

#pragma unroll
    for (int mi = 0; mi < 2; ++mi) {
      const __bf16* ar = &Al[(wm * 32 + mi * 16 + (lane & 15)) * 32 + h * 8];
      v16bf af = frag_from2(ar, ar + 16);
#pragma unroll
      for (int ni = 0; ni < 2; ++ni) {
        const __bf16* br = &Bl[(wn * 32 + ni * 16 + (lane & 15)) * 32 + h * 8];
        v16bf bfr = frag_from2(br, br + 16);
        acc[mi][ni] = wmma_bf16(af, bfr, acc[mi][ni]);
      }
    }
  }

  // epilogue: C-tile layout -> lanes 0-15: M=r, lanes 16-31: M=r+8; N=lane%16
#pragma unroll
  for (int mi = 0; mi < 2; ++mi)
#pragma unroll
    for (int ni = 0; ni < 2; ++ni) {
      int n = n0 + wn * 32 + ni * 16 + (lane & 15);
      float bv = bias ? bias[n] : 0.f;
#pragma unroll
      for (int r = 0; r < 8; ++r) {
        int m = m0 + wm * 32 + mi * 16 + r + 8 * h;
        float v = acc[mi][ni][r] + bv;
        size_t o = (size_t)m * Nout + n;
        if (outF) outF[o] = v;
        if (outB) outB[o] = (__bf16)v;
      }
    }
}

// ---------------------------------------------------------------------------
// Attention pass 1: per-row max & sum-exp of S = Q K^T (C=256).
// Grid: (Npts/128, B). Each wave owns a 16-row tile; Q rows held in regs.
// ---------------------------------------------------------------------------
__global__ __launch_bounds__(256)
void k_attn_pass1(const __bf16* __restrict__ Q, const __bf16* __restrict__ Kv,
                  float* __restrict__ rowmax, float* __restrict__ rowrz, int Npts) {
  const int C = 256;
  __shared__ __bf16 kT[16 * 256];   // one 16-key tile, row-major (8KB)
  const int b = blockIdx.y;
  const int rowbase = blockIdx.x * 128;
  const __bf16* q = Q + (size_t)b * Npts * C;
  const __bf16* kp = Kv + (size_t)b * Npts * C;
  const int tid = threadIdx.x, lane = tid & 31, wid = tid >> 5, h = lane >> 4;

  // preload this wave's 16x256 Q block as 8 A-fragments
  const __bf16* qrow = q + (size_t)(rowbase + wid * 16 + (lane & 15)) * C;
  v16bf qa[8];
#pragma unroll
  for (int ks = 0; ks < 8; ++ks)
    qa[ks] = frag_from2(qrow + ks * 32 + h * 8, qrow + ks * 32 + 16 + h * 8);

  float mx[8], sm[8];
#pragma unroll
  for (int i = 0; i < 8; ++i) { mx[i] = -3.0e38f; sm[i] = 0.f; }

  for (int nt = 0; nt < Npts / 16; ++nt) {
    __syncthreads();
    for (int c = tid; c < 512; c += 256) {    // 16 rows x 512B
      int r = c >> 5, p = c & 31;
      async_copy16(&kT[r * 256 + p * 8], kp + (size_t)(nt * 16 + r) * C + p * 8);
    }
    async_wait0();
    __syncthreads();

    v8f acc = {};
#pragma unroll
    for (int ks = 0; ks < 8; ++ks) {
      const __bf16* kr = &kT[(lane & 15) * 256 + ks * 32 + h * 8];
      acc = wmma_bf16(qa[ks], frag_from2(kr, kr + 16), acc);
    }
    // online softmax stats per row (rows live in lane halves of each VGPR slot)
#pragma unroll
    for (int i = 0; i < 8; ++i) {
      float v = acc[i];
      float tmax = v;
      for (int m = 8; m >= 1; m >>= 1) tmax = fmaxf(tmax, __shfl_xor(tmax, m, 32));
      float p = __expf(v - tmax);
      for (int m = 8; m >= 1; m >>= 1) p += __shfl_xor(p, m, 32);
      float mnew = fmaxf(mx[i], tmax);
      sm[i] = sm[i] * __expf(mx[i] - mnew) + p * __expf(tmax - mnew);
      mx[i] = mnew;
    }
  }
  if ((lane & 15) == 0) {
#pragma unroll
    for (int i = 0; i < 8; ++i) {
      int row = rowbase + wid * 16 + i + 8 * h;
      rowmax[(size_t)b * Npts + row] = mx[i];
      rowrz[(size_t)b * Npts + row] = 1.0f / sm[i];
    }
  }
}

// ---------------------------------------------------------------------------
// Attention pass 2: colacc[b,n] += sum_m exp(S[m,n]-max_m)/Z_m (f32 atomics).
// ---------------------------------------------------------------------------
__global__ __launch_bounds__(256)
void k_attn_pass2(const __bf16* __restrict__ Q, const __bf16* __restrict__ Kv,
                  const float* __restrict__ rowmax, const float* __restrict__ rowrz,
                  float* __restrict__ colacc, int Npts) {
  const int C = 256;
  __shared__ __bf16 kT[16 * 256];
  const int b = blockIdx.y;
  const int rowbase = blockIdx.x * 128;
  const __bf16* q = Q + (size_t)b * Npts * C;
  const __bf16* kp = Kv + (size_t)b * Npts * C;
  const int tid = threadIdx.x, lane = tid & 31, wid = tid >> 5, h = lane >> 4;

  const __bf16* qrow = q + (size_t)(rowbase + wid * 16 + (lane & 15)) * C;
  v16bf qa[8];
#pragma unroll
  for (int ks = 0; ks < 8; ++ks)
    qa[ks] = frag_from2(qrow + ks * 32 + h * 8, qrow + ks * 32 + 16 + h * 8);

  float rm[8], rz[8];
#pragma unroll
  for (int i = 0; i < 8; ++i) {
    int row = rowbase + wid * 16 + i + 8 * h;
    rm[i] = rowmax[(size_t)b * Npts + row];
    rz[i] = rowrz[(size_t)b * Npts + row];
  }

  for (int nt = 0; nt < Npts / 16; ++nt) {
    __syncthreads();
    for (int c = tid; c < 512; c += 256) {
      int r = c >> 5, p = c & 31;
      async_copy16(&kT[r * 256 + p * 8], kp + (size_t)(nt * 16 + r) * C + p * 8);
    }
    async_wait0();
    __syncthreads();

    v8f acc = {};
#pragma unroll
    for (int ks = 0; ks < 8; ++ks) {
      const __bf16* kr = &kT[(lane & 15) * 256 + ks * 32 + h * 8];
      acc = wmma_bf16(qa[ks], frag_from2(kr, kr + 16), acc);
    }
    float colpart = 0.f;
#pragma unroll
    for (int i = 0; i < 8; ++i)
      colpart += __expf(acc[i] - rm[i]) * rz[i];
    colpart += __shfl_xor(colpart, 16, 32);   // combine lane halves (M 0-7 + 8-15)
    if (lane < 16)
      atomicAdd(&colacc[(size_t)b * Npts + nt * 16 + lane], colpart);
  }
}

// ---------------------------------------------------------------------------
// vs = v * colacc/(eps + colacc), write bf16 for the output projection GEMM.
// ---------------------------------------------------------------------------
__global__ void k_scale_v(const float* __restrict__ vf, const float* __restrict__ colacc,
                          __bf16* __restrict__ vs, int total, int C) {
  int idx = blockIdx.x * blockDim.x + threadIdx.x;
  if (idx >= total) return;
  float t = colacc[idx / C];
  float col = t / (1e-9f + t);
  vs[idx] = (__bf16)(vf[idx] * col);
}

// ---------------------------------------------------------------------------
// Host driver
// ---------------------------------------------------------------------------
extern "C" void kernel_launch(void* const* d_in, const int* in_sizes, int n_in,
                              void* d_out, int out_size, void* d_ws, size_t ws_size,
                              hipStream_t stream) {
  (void)in_sizes; (void)n_in; (void)out_size; (void)ws_size;
  const int B = 2, N = 4096, C = 256, F = 1024, L = 4;
  const int BN = B * N, NC = N * C;

  const float* inpts = (const float*)d_in[0];
  const float* w0    = (const float*)d_in[1];
  const float* b0    = (const float*)d_in[2];
  const float* ln0s  = (const float*)d_in[3];
  const float* ln0b  = (const float*)d_in[4];
  const float* w1    = (const float*)d_in[5];
  const float* b1    = (const float*)d_in[6];
  const float* ln1s  = (const float*)d_in[7];
  const float* ln1b  = (const float*)d_in[8];
  const float* wq    = (const float*)d_in[9];
  const float* bq    = (const float*)d_in[10];
  const float* wk    = (const float*)d_in[11];
  const float* bk    = (const float*)d_in[12];
  const float* wv    = (const float*)d_in[13];
  const float* bv    = (const float*)d_in[14];
  const float* wo    = (const float*)d_in[15];
  const float* bo    = (const float*)d_in[16];
  const float* alns  = (const float*)d_in[17];
  const float* alnb  = (const float*)d_in[18];
  const float* wf    = (const float*)d_in[19];
  const float* bf    = (const float*)d_in[20];
  float* out = (float*)d_out;

  // ---- carve workspace ----
  char* ws = (char*)d_ws;
  size_t off = 0;
  auto carve = [&](size_t bytes) -> void* {
    void* p = ws + off;
    off += (bytes + 255) & ~(size_t)255;
    return p;
  };
  float*  Xf   = (float*)carve((size_t)BN * C * 4);
  __bf16* Xb   = (__bf16*)carve((size_t)BN * C * 2);
  float*  tmp  = (float*)carve((size_t)BN * C * 4);
  __bf16* qb   = (__bf16*)carve((size_t)BN * C * 2);
  __bf16* kb   = (__bf16*)carve((size_t)BN * C * 2);
  float*  vf   = (float*)carve((size_t)BN * C * 4);
  __bf16* vs   = (__bf16*)carve((size_t)BN * C * 2);
  __bf16* cat  = (__bf16*)carve((size_t)BN * 4 * C * 2);
  float*  rowm = (float*)carve((size_t)BN * 4);
  float*  rowz = (float*)carve((size_t)BN * 4);
  float*  cacc = (float*)carve((size_t)BN * 4);
  float*  mean = (float*)carve((size_t)B * C * 4);
  float*  rstd = (float*)carve((size_t)B * C * 4);
  __bf16* w1t  = (__bf16*)carve((size_t)C * C * 2);
  __bf16* wqt  = (__bf16*)carve((size_t)L * C * C * 2);
  __bf16* wkt  = (__bf16*)carve((size_t)L * C * C * 2);
  __bf16* wvt  = (__bf16*)carve((size_t)L * C * C * 2);
  __bf16* wot  = (__bf16*)carve((size_t)L * C * C * 2);
  __bf16* wft  = (__bf16*)carve((size_t)4 * C * F * 2);

  auto g1 = [](int n) { return dim3((unsigned)((n + 255) / 256)); };

  // ---- weight convert + transpose (bf16, K-major per output column) ----
  k_convT<<<g1(C * C), 256, 0, stream>>>(w1, w1t, C, C);
  for (int i = 0; i < L; ++i) {
    k_convT<<<g1(C * C), 256, 0, stream>>>(wq + (size_t)i * C * C, wqt + (size_t)i * C * C, C, C);
    k_convT<<<g1(C * C), 256, 0, stream>>>(wk + (size_t)i * C * C, wkt + (size_t)i * C * C, C, C);
    k_convT<<<g1(C * C), 256, 0, stream>>>(wv + (size_t)i * C * C, wvt + (size_t)i * C * C, C, C);
    k_convT<<<g1(C * C), 256, 0, stream>>>(wo + (size_t)i * C * C, wot + (size_t)i * C * C, C, C);
  }
  k_convT<<<g1(4 * C * F), 256, 0, stream>>>(wf, wft, 4 * C, F);

  // ---- embed + point-LN ----
  k_embed<<<g1(BN * C), 256, 0, stream>>>(inpts, w0, b0, tmp, BN * C, C);
  k_colstats<<<B * C, 256, 0, stream>>>(tmp, mean, rstd, N, C);
  k_ln_apply<<<g1(BN * C), 256, 0, stream>>>(tmp, nullptr, mean, rstd, ln0s, ln0b,
                                             Xf, Xb, nullptr, 0, BN * C, NC, C, 0);
  // ---- conv1 + point-LN ----
  k_gemm_bf16<<<dim3(BN / 128, C / 64), 256, 0, stream>>>(Xb, w1t, b1, tmp, nullptr, BN, C, C);
  k_colstats<<<B * C, 256, 0, stream>>>(tmp, mean, rstd, N, C);
  k_ln_apply<<<g1(BN * C), 256, 0, stream>>>(tmp, nullptr, mean, rstd, ln1s, ln1b,
                                             Xf, Xb, nullptr, 0, BN * C, NC, C, 0);

  // ---- attention layers ----
  for (int i = 0; i < L; ++i) {
    const size_t wOff = (size_t)i * C * C;
    k_gemm_bf16<<<dim3(BN / 128, C / 64), 256, 0, stream>>>(Xb, wqt + wOff, bq + (size_t)i * C,
                                                            nullptr, qb, BN, C, C);
    k_gemm_bf16<<<dim3(BN / 128, C / 64), 256, 0, stream>>>(Xb, wkt + wOff, bk + (size_t)i * C,
                                                            nullptr, kb, BN, C, C);
    k_gemm_bf16<<<dim3(BN / 128, C / 64), 256, 0, stream>>>(Xb, wvt + wOff, bv + (size_t)i * C,
                                                            vf, nullptr, BN, C, C);
    k_attn_pass1<<<dim3(N / 128, B), 256, 0, stream>>>(qb, kb, rowm, rowz, N);
    k_zero<<<g1(BN), 256, 0, stream>>>(cacc, BN);
    k_attn_pass2<<<dim3(N / 128, B), 256, 0, stream>>>(qb, kb, rowm, rowz, cacc, N);
    k_scale_v<<<g1(BN * C), 256, 0, stream>>>(vf, cacc, vs, BN * C, C);
    k_gemm_bf16<<<dim3(BN / 128, C / 64), 256, 0, stream>>>(vs, wot + wOff, bo + (size_t)i * C,
                                                            tmp, nullptr, BN, C, C);
    k_colstats<<<B * C, 256, 0, stream>>>(tmp, mean, rstd, N, C);
    // relu(LN(.)) + residual; also write the concat slice for the final GEMM
    k_ln_apply<<<g1(BN * C), 256, 0, stream>>>(tmp, Xf, mean, rstd,
                                               alns + (size_t)i * C, alnb + (size_t)i * C,
                                               Xf, Xb, cat + (size_t)i * C, 4 * C,
                                               BN * C, NC, C, 1);
  }

  // ---- final projection [BN, 4C] @ [4C, F] ----
  k_gemm_bf16<<<dim3(BN / 128, F / 64), 256, 0, stream>>>(cat, wft, bf, out, nullptr,
                                                          BN, 4 * C, F);
}
